// DAM_60172491817464
// MI455X (gfx1250) — compile-verified, matrix operand order checked by reading, not run on hardware
//
#include <hip/hip_runtime.h>
#include <math.h>

// Problem constants (match reference)
#define BATCH 4
#define CH    64
#define HH    128
#define WW    128
#define KK    9
#define HD    64
#define HWSZ  (HH*WW)          // 16384 = 2^14
#define NPIX  (BATCH*HWSZ)     // 65536
#define NCAT  160              // [dx(9)+dy(9) pad->32 | Qt(64) | Q(64)]
#define KFIN  128              // [sbar(64) | Q(64)]

typedef __attribute__((ext_vector_type(16))) _Float16 v16h;
typedef __attribute__((ext_vector_type(8)))  _Float16 v8h;
typedef __attribute__((ext_vector_type(2)))  _Float16 v2h;
typedef __attribute__((ext_vector_type(8)))  float    v8f;
typedef __attribute__((ext_vector_type(4)))  int      v4i;

// workspace layout (bytes); total ~80 MB, fits comfortably in 192MB L2
#define OFF_XTF  0ull          // NPIX*64 f32  = 16 MB   (NHWC f32, for bilinear gather)
#define OFF_XTH  16777216ull   // NPIX*64 f16  = 8 MB    (NHWC f16, WMMA A operand)
#define OFF_P    25165824ull   // NPIX*160 f32 = 40 MB   (offsets | Qt | Q per pixel)
#define OFF_SQ   67108864ull   // NPIX*128 f16 = 16 MB   ([sbar | Q] per pixel, f16)
#define OFF_WCAT 83886080ull   // 160*64 f16
#define OFF_BCAT 83906560ull   // 160 f32
#define OFF_WFIN 83907200ull   // 64*128 f16
#define OFF_BFIN 83923584ull   // 64 f32

// ---- async global->LDS staging (ASYNCcnt path), with VGPR-copy fallback ----
#if defined(__has_builtin)
#if __has_builtin(__builtin_amdgcn_global_load_async_to_lds_b128) && \
    __has_builtin(__builtin_amdgcn_s_wait_asynccnt)
#define HAVE_ASYNC_LDS 1
#endif
#endif

__device__ inline void stage_to_lds(_Float16* lds_dst, const _Float16* gsrc,
                                    int nbytes, int t, int nthreads) {
#if defined(HAVE_ASYNC_LDS)
  __attribute__((address_space(3))) v4i* lp =
      (__attribute__((address_space(3))) v4i*)lds_dst;
  __attribute__((address_space(1))) v4i* gp =
      (__attribute__((address_space(1))) v4i*)(_Float16*)gsrc;
  for (int i = t; i < nbytes / 16; i += nthreads)
    __builtin_amdgcn_global_load_async_to_lds_b128(gp + i, lp + i, 0, 0);
  __builtin_amdgcn_s_wait_asynccnt(0);
#else
  const uint4* src = (const uint4*)gsrc;
  uint4* dst = (uint4*)lds_dst;
  for (int i = t; i < nbytes / 16; i += nthreads) dst[i] = src[i];
#endif
}

// Load one 16x32 f16 WMMA operand tile from a row-major matrix (rows = M for A,
// rows = N (i.e. B^T) for B). Matches CDNA5 16-bit A layout:
//   lane l<16 : row l,    K = kk+{0..7} and kk+{16..23}
//   lane l>=16: row l-16, K = kk+{8..15} and kk+{24..31}
// Two aligned 16B loads per lane.
__device__ inline v16h load_ab_tile(const _Float16* base, int rowStride, int lane, int kk) {
  int r  = lane & 15;
  int k0 = kk + ((lane & 16) ? 8 : 0);
  const _Float16* p = base + (size_t)r * rowStride + k0;
  v8h lo = *(const v8h*)(p);
  v8h hi = *(const v8h*)(p + 16);
  v16h o;
#pragma unroll
  for (int i = 0; i < 8; ++i) { o[i] = lo[i]; o[i+8] = hi[i]; }
  return o;
}

// ---------------------------------------------------------------- kernel 0
// Fold the whole network into 2 small GEMM weight sets:
//  Wcat (160x64): rows 0..17 off_w, 18..31 zero pad,
//                 rows 32..95  Mc = diag(k_dw) k_pw^T q_pw diag(q_dw)  (Qt = Mc x + cb)
//                 rows 96..159 Wq' = q_pw diag(q_dw)                   (Q  = Wq' x + q_pb)
//  Wfin (64x128): cols 0..63  Wo2 = alpha * o_pw diag(o_dw) v_pw diag(v_dw)
//                 cols 64..127 Wo1 = o_pw diag(o_dw)
//  (k_pb drops out: it shifts all K scores equally -> cancels in softmax)
__global__ void dam_prep_kernel(const float* __restrict__ alpha,
                                const float* __restrict__ off_w, const float* __restrict__ off_b,
                                const float* __restrict__ q_dw,  const float* __restrict__ q_pw,
                                const float* __restrict__ q_pb,
                                const float* __restrict__ k_dw,  const float* __restrict__ k_pw,
                                const float* __restrict__ v_dw,  const float* __restrict__ v_pw,
                                const float* __restrict__ v_pb,
                                const float* __restrict__ o_dw,  const float* __restrict__ o_pw,
                                const float* __restrict__ o_pb,
                                _Float16* __restrict__ Wcat, float* __restrict__ bcat,
                                _Float16* __restrict__ Wfin, float* __restrict__ bfin) {
  const int t = threadIdx.x;
  const float al = alpha[0];
  for (int e = t; e < NCAT*CH; e += 256) {
    int r = e >> 6, c = e & 63;
    float v = 0.f;
    if (r < 18) v = off_w[r*CH + c];
    else if (r < 32) v = 0.f;
    else if (r < 96) {
      int i = r - 32; float s = 0.f;
      for (int j = 0; j < HD; ++j) s += k_pw[j*CH + i] * q_pw[j*CH + c];
      v = k_dw[i] * q_dw[c] * s;
    } else {
      int j = r - 96;
      v = q_pw[j*CH + c] * q_dw[c];
    }
    Wcat[e] = (_Float16)v;
  }
  for (int r = t; r < NCAT; r += 256) {
    float v = 0.f;
    if (r < 18) v = off_b[r];
    else if (r < 32) v = 0.f;
    else if (r < 96) {
      int i = r - 32; float s = 0.f;
      for (int j = 0; j < HD; ++j) s += k_pw[j*CH + i] * q_pb[j];
      v = k_dw[i] * s;
    } else v = q_pb[r - 96];
    bcat[r] = v;
  }
  for (int e = t; e < HD*KFIN; e += 256) {
    int o = e >> 7, k = e & 127;
    float v;
    if (k < 64) {
      float s = 0.f;
      for (int j = 0; j < HD; ++j) s += o_pw[o*HD + j] * o_dw[j] * v_pw[j*CH + k];
      v = al * v_dw[k] * s;
    } else {
      v = o_pw[o*HD + (k-64)] * o_dw[k-64];
    }
    Wfin[e] = (_Float16)v;
  }
  for (int o = t; o < CH; o += 256) {
    float s = 0.f;
    for (int j = 0; j < HD; ++j) s += o_pw[o*HD + j] * o_dw[j] * v_pb[j];
    bfin[o] = o_pb[o] + al * s;
  }
}

// ---------------------------------------------------------------- kernel 1
// NCHW -> NHWC transpose (f32 + f16 copies), 64x64 tile via LDS.
__global__ void dam_transpose_kernel(const float* __restrict__ x,
                                     float* __restrict__ xtf, _Float16* __restrict__ xth) {
  __shared__ float tile[64][65];
  const int p0 = blockIdx.x * 64;           // global pixel base (b*HW + pix)
  const int b  = p0 >> 14;
  const int pixbase = p0 & (HWSZ - 1);
  const int tx = threadIdx.x & 63;
  const int ty = threadIdx.x >> 6;          // 0..3
  const float* xb = x + (size_t)b * CH * HWSZ;
#pragma unroll
  for (int i = 0; i < 16; ++i) {
    int c = i*4 + ty;
    tile[c][tx] = xb[(size_t)c * HWSZ + pixbase + tx];
  }
  __syncthreads();
#pragma unroll
  for (int i = 0; i < 16; ++i) {
    int pl = i*4 + ty;
    float v = tile[tx][pl];
    size_t idx = (size_t)(p0 + pl) * CH + tx;
    xtf[idx] = v;
    xth[idx] = (_Float16)v;
  }
}

// ---------------------------------------------------------------- kernel 2
// P[pix][0:160] = xth[pix][0:64] @ Wcat^T + bcat   (offsets | Qt | Q)
// 128 pixels per block, 8 waves, each wave: 16 pixels x 10 N-tiles x (K=64 -> 2 wmma)
__global__ void dam_gemm1_kernel(const _Float16* __restrict__ xth,
                                 const _Float16* __restrict__ Wcat,
                                 const float* __restrict__ bcat,
                                 float* __restrict__ P) {
  __shared__ __align__(16) _Float16 wlds[NCAT*CH];   // 20 KB
  const int t = threadIdx.x;
  stage_to_lds(wlds, Wcat, NCAT*CH*2, t, 256);
  __syncthreads();
  const int wv = t >> 5, lane = t & 31;
  const int m0 = blockIdx.x * 128 + wv * 16;
  const _Float16* abase = xth + (size_t)m0 * CH;
  v16h A0 = load_ab_tile(abase, CH, lane, 0);
  v16h A1 = load_ab_tile(abase, CH, lane, 32);
  const int mrow = (lane & 16) ? 8 : 0;
  const int nc   = lane & 15;
#pragma unroll
  for (int n = 0; n < 10; ++n) {
    v16h B0 = load_ab_tile(wlds + n*16*CH, CH, lane, 0);
    v16h B1 = load_ab_tile(wlds + n*16*CH, CH, lane, 32);
    v8f acc = {};
    acc = __builtin_amdgcn_wmma_f32_16x16x32_f16(false, A0, false, B0, (short)0, acc, false, false);
    acc = __builtin_amdgcn_wmma_f32_16x16x32_f16(false, A1, false, B1, (short)0, acc, false, false);
    const int ncol = n*16 + nc;
    const float bb = bcat[ncol];
#pragma unroll
    for (int r = 0; r < 8; ++r) {
      size_t pxl = (size_t)(m0 + mrow + r);
      P[pxl*NCAT + ncol] = acc[r] + bb;
    }
  }
}

// ---------------------------------------------------------------- kernel 3
// One wave per pixel. Lane l owns channels {2l, 2l+1} so every bilinear corner
// is a single aligned global_load_b64. Scores via Qt dot + wave reduction,
// softmax over 9 taps, sbar accumulation. Writes [sbar | Q] as f16 pairs.
__device__ inline float2 dam_gather2(const float* __restrict__ xb, int yi, int xi, int lane) {
  bool ok = (yi >= 0) & (yi < HH) & (xi >= 0) & (xi < WW);
  int yc = yi < 0 ? 0 : (yi > HH-1 ? HH-1 : yi);
  int xc = xi < 0 ? 0 : (xi > WW-1 ? WW-1 : xi);
  const float2* rp = (const float2*)(xb + (size_t)(yc * WW + xc) * CH);
  float2 v = rp[lane];
  float m = ok ? 1.0f : 0.0f;
  v.x *= m; v.y *= m;
  return v;
}

__global__ void dam_attn_kernel(const float* __restrict__ xtf,
                                const float* __restrict__ P,
                                _Float16* __restrict__ SQ) {
  const int t = threadIdx.x;
  const int wv = t >> 5, lane = t & 31;
  const int p  = blockIdx.x * 8 + wv;
  const int b  = p >> 14;
  const int pix = p & (HWSZ - 1);
  const int yy = pix >> 7;
  const int xx = pix & (WW - 1);
  const float* prow = P + (size_t)p * NCAT;
  const float* xb   = xtf + (size_t)b * HWSZ * CH;
  const float2 qt = ((const float2*)(prow + 32))[lane];
  float s0[KK], s1[KK], sc[KK];
#pragma unroll
  for (int k = 0; k < KK; ++k) {
    float dxk = prow[k];
    float dyk = prow[KK + k];
    float gx = 2.0f * ((float)xx + dxk) / (float)(WW - 1) - 1.0f;
    float gy = 2.0f * ((float)yy + dyk) / (float)(HH - 1) - 1.0f;
    float ix = ((gx + 1.0f) * (float)WW - 1.0f) * 0.5f;
    float iy = ((gy + 1.0f) * (float)HH - 1.0f) * 0.5f;
    float fx = floorf(ix), fy = floorf(iy);
    float wx = ix - fx, wy = iy - fy;
    int x0 = (int)fx, y0 = (int)fy;
    float2 va = dam_gather2(xb, y0,   x0,   lane);
    float2 vb = dam_gather2(xb, y0,   x0+1, lane);
    float2 vc = dam_gather2(xb, y0+1, x0,   lane);
    float2 vd = dam_gather2(xb, y0+1, x0+1, lane);
    float w00 = (1.f-wy)*(1.f-wx), w01 = (1.f-wy)*wx, w10 = wy*(1.f-wx), w11 = wy*wx;
    float t0 = w00*va.x + w01*vb.x + w10*vc.x + w11*vd.x;
    float t1 = w00*va.y + w01*vb.y + w10*vc.y + w11*vd.y;
    s0[k] = t0; s1[k] = t1;
    float d = qt.x*t0 + qt.y*t1;
#pragma unroll
    for (int off = 16; off > 0; off >>= 1) d += __shfl_xor(d, off, 32);
    sc[k] = d;
  }
  float mx = sc[0];
#pragma unroll
  for (int k = 1; k < KK; ++k) mx = fmaxf(mx, sc[k]);
  float se = 0.f;
#pragma unroll
  for (int k = 0; k < KK; ++k) { sc[k] = expf(sc[k] - mx); se += sc[k]; }
  const float inv = 1.f / se;
  float sb0 = 0.f, sb1 = 0.f;
#pragma unroll
  for (int k = 0; k < KK; ++k) { float a = sc[k]*inv; sb0 += a*s0[k]; sb1 += a*s1[k]; }
  _Float16* srow = SQ + (size_t)p * KFIN;
  v2h pk; pk[0] = (_Float16)sb0; pk[1] = (_Float16)sb1;
  *(v2h*)(srow + 2*lane) = pk;
  const float2 qn = ((const float2*)(prow + 96))[lane];
  v2h qk; qk[0] = (_Float16)qn.x; qk[1] = (_Float16)qn.y;
  *(v2h*)(srow + 64 + 2*lane) = qk;
}

// ---------------------------------------------------------------- kernel 4
// out = SQ[pix][0:128] @ Wfin^T + bfin, scattered back to NCHW f32.
__global__ void dam_gemm2_kernel(const _Float16* __restrict__ SQ,
                                 const _Float16* __restrict__ Wfin,
                                 const float* __restrict__ bfin,
                                 float* __restrict__ out) {
  __shared__ __align__(16) _Float16 wlds[HD*KFIN];   // 16 KB
  const int t = threadIdx.x;
  stage_to_lds(wlds, Wfin, HD*KFIN*2, t, 256);
  __syncthreads();
  const int wv = t >> 5, lane = t & 31;
  const int m0 = blockIdx.x * 128 + wv * 16;
  const _Float16* abase = SQ + (size_t)m0 * KFIN;
  v16h A[4];
#pragma unroll
  for (int s = 0; s < 4; ++s) A[s] = load_ab_tile(abase, KFIN, lane, s*32);
  const int mrow = (lane & 16) ? 8 : 0;
  const int nc   = lane & 15;
#pragma unroll
  for (int n = 0; n < 4; ++n) {
    v8f acc = {};
#pragma unroll
    for (int s = 0; s < 4; ++s) {
      v16h Bv = load_ab_tile(wlds + n*16*KFIN, KFIN, lane, s*32);
      acc = __builtin_amdgcn_wmma_f32_16x16x32_f16(false, A[s], false, Bv, (short)0, acc, false, false);
    }
    const int c = n*16 + nc;
    const float bb = bfin[c];
#pragma unroll
    for (int r = 0; r < 8; ++r) {
      int pixel = m0 + mrow + r;
      int b = pixel >> 14;
      int pix = pixel & (HWSZ - 1);
      out[((size_t)b*CH + c)*HWSZ + pix] = acc[r] + bb;
    }
  }
}

// ----------------------------------------------------------------
extern "C" void kernel_launch(void* const* d_in, const int* in_sizes, int n_in,
                              void* d_out, int out_size, void* d_ws, size_t ws_size,
                              hipStream_t stream) {
  (void)in_sizes; (void)n_in; (void)out_size; (void)ws_size;
  const float* x     = (const float*)d_in[0];
  const float* alpha = (const float*)d_in[1];
  const float* off_w = (const float*)d_in[2];
  const float* off_b = (const float*)d_in[3];
  const float* q_dw  = (const float*)d_in[4];
  const float* q_pw  = (const float*)d_in[5];
  const float* q_pb  = (const float*)d_in[6];
  const float* k_dw  = (const float*)d_in[7];
  const float* k_pw  = (const float*)d_in[8];
  // d_in[9] = k_pb: provably cancels in softmax, not needed
  const float* v_dw  = (const float*)d_in[10];
  const float* v_pw  = (const float*)d_in[11];
  const float* v_pb  = (const float*)d_in[12];
  const float* o_dw  = (const float*)d_in[13];
  const float* o_pw  = (const float*)d_in[14];
  const float* o_pb  = (const float*)d_in[15];

  char* ws = (char*)d_ws;
  float*    xtf  = (float*)(ws + OFF_XTF);
  _Float16* xth  = (_Float16*)(ws + OFF_XTH);
  float*    P    = (float*)(ws + OFF_P);
  _Float16* SQ   = (_Float16*)(ws + OFF_SQ);
  _Float16* Wcat = (_Float16*)(ws + OFF_WCAT);
  float*    bcat = (float*)(ws + OFF_BCAT);
  _Float16* Wfin = (_Float16*)(ws + OFF_WFIN);
  float*    bfin = (float*)(ws + OFF_BFIN);

  hipLaunchKernelGGL(dam_prep_kernel, dim3(1), dim3(256), 0, stream,
                     alpha, off_w, off_b, q_dw, q_pw, q_pb, k_dw, k_pw,
                     v_dw, v_pw, v_pb, o_dw, o_pw, o_pb, Wcat, bcat, Wfin, bfin);
  hipLaunchKernelGGL(dam_transpose_kernel, dim3(NPIX/64), dim3(256), 0, stream, x, xtf, xth);
  hipLaunchKernelGGL(dam_gemm1_kernel, dim3(NPIX/128), dim3(256), 0, stream, xth, Wcat, bcat, P);
  hipLaunchKernelGGL(dam_attn_kernel, dim3(NPIX/8), dim3(256), 0, stream, xtf, P, SQ);
  hipLaunchKernelGGL(dam_gemm2_kernel, dim3(NPIX/128), dim3(256), 0, stream, SQ, Wfin, bfin, (float*)d_out);
}
